// TiedSingleHeadAttention_28355374088947
// MI455X (gfx1250) — matrix operand order repaired and checked by reading, not compiled, not run on hardware
//
#include <hip/hip_runtime.h>
#include <hip/hip_bf16.h>
#include <math.h>

typedef _Float16 v16h __attribute__((ext_vector_type(16)));
typedef _Float16 v8h  __attribute__((ext_vector_type(8)));
typedef _Float16 v2h  __attribute__((ext_vector_type(2)));
typedef float    v8f  __attribute__((ext_vector_type(8)));

#define BB 8
#define TT 2048
#define DD 128
#define RR 64

#define XK_LD  136   // halfs, padded stride, key tile [64][128]
#define XKT_LD 72    // halfs, padded stride, transposed key tile [128][64]
#define P_LD   72    // halfs, padded stride, P staging [128][64]
#define QH_LD  136   // halfs, padded stride, Q staging [128][128]

#define BUF_STRIDE 35840   // Xk (17408) + XkT (18432) per ping-pong buffer
#define P_OFF      71680   // 2 * BUF_STRIDE
#define SMEM_BYTES 90112   // P_OFF + 18432

__device__ __forceinline__ v16h ld16(const _Float16* p0, const _Float16* p1) {
  v8h a = *(const v8h*)p0;
  v8h b = *(const v8h*)p1;
  return __builtin_shufflevector(a, b, 0,1,2,3,4,5,6,7,8,9,10,11,12,13,14,15);
}

__device__ __forceinline__ v8f zero8() {
  v8f z = {0.f,0.f,0.f,0.f,0.f,0.f,0.f,0.f};
  return z;
}

__device__ __forceinline__ v8f wmma_f16(v16h a, v16h b, v8f c) {
  // D = A(16x32 f16) * B(32x16 f16) + C(16x16 f32)
  return __builtin_amdgcn_wmma_f32_16x16x32_f16(false, a, false, b, (short)0, c,
                                                false, false);
}

// Async copy 16B global -> LDS (ASYNCcnt-tracked; drained before the barrier).
__device__ __forceinline__ void async_ld128(const _Float16* gsrc, _Float16* ldst) {
  unsigned d = (unsigned)(uintptr_t)(void*)ldst;  // flat->LDS offset (low 32b)
  asm volatile("global_load_async_to_lds_b128 %0, %1, off"
               :: "v"(d), "v"(gsrc) : "memory");
}

__device__ __forceinline__ void async_drain() {
  asm volatile("s_wait_asynccnt 0" ::: "memory");
}

// Each thread owns key-pair (2p, 2p+1) x 16-dim segment cs.
__device__ __forceinline__ void tile_async_rowmajor(_Float16* Xk, const _Float16* Xb,
                                                    int k0, int p, int cs) {
  const _Float16* s0 = Xb + (size_t)(k0 + 2 * p) * DD + cs;
  async_ld128(s0,          Xk + (2 * p) * XK_LD + cs);
  async_ld128(s0 + 8,      Xk + (2 * p) * XK_LD + cs + 8);
  async_ld128(s0 + DD,     Xk + (2 * p + 1) * XK_LD + cs);
  async_ld128(s0 + DD + 8, Xk + (2 * p + 1) * XK_LD + cs + 8);
}

__device__ __forceinline__ void tile_load_regs(v8h st[4], const _Float16* Xb,
                                               int k0, int p, int cs) {
  const _Float16* s0 = Xb + (size_t)(k0 + 2 * p) * DD + cs;
  st[0] = *(const v8h*)(s0);
  st[1] = *(const v8h*)(s0 + 8);
  st[2] = *(const v8h*)(s0 + DD);
  st[3] = *(const v8h*)(s0 + DD + 8);
}

// Transposed copy only: pack two adjacent keys into one dword store.
__device__ __forceinline__ void tile_store_tr(_Float16* XkT, const v8h st[4],
                                              int p, int cs) {
  #pragma unroll
  for (int i = 0; i < 16; ++i) {
    v2h pr;
    pr[0] = (i < 8) ? st[0][i] : st[1][i - 8];
    pr[1] = (i < 8) ? st[2][i] : st[3][i - 8];
    *(v2h*)(XkT + (cs + i) * XKT_LD + 2 * p) = pr;
  }
}

// ---------------------------------------------------------------------------
// Pass A: S = W W^T / sqrt(d*r) -> f16; zero per-batch trace accumulators
// ---------------------------------------------------------------------------
__global__ void k_prep_s(const float* __restrict__ W, _Float16* __restrict__ Sh,
                         float* __restrict__ trace) {
  const int tid = threadIdx.x;
  if (tid < BB) trace[tid] = 0.0f;
  const float sc = 0.011048543456039805f;  // 1/sqrt(128*64)
  for (int idx = tid; idx < DD * DD; idx += 256) {
    const int i = idx >> 7, j = idx & 127;
    float acc = 0.f;
    for (int r = 0; r < RR; ++r) acc += W[i * RR + r] * W[j * RR + r];
    Sh[idx] = (_Float16)(acc * sc);
  }
}

// ---------------------------------------------------------------------------
// Pass B: X -> f16; trace_b += sum_rows ||W^T x||^2 / sqrt(d*r)
// (tr(X S X^T) = sum_t x^T S x = sum_t |W^T x_t|^2 / sqrt(d*r))
// ---------------------------------------------------------------------------
__global__ void k_prep_x(const float* __restrict__ X, const float* __restrict__ W,
                         _Float16* __restrict__ Xh, float* __restrict__ trace) {
  __shared__ float Wl[DD * RR];  // 32 KB
  __shared__ float red[256];
  const int tid = threadIdx.x;
  for (int i = tid; i < DD * RR; i += 256) Wl[i] = W[i];
  __syncthreads();

  const long row = (long)blockIdx.x * 256 + tid;  // 0..16383
  const int b = (int)(row >> 11);
  const float* x = X + row * DD;

  float dj[RR];
  #pragma unroll
  for (int j = 0; j < RR; ++j) dj[j] = 0.f;

  for (int e = 0; e < DD; ++e) {
    const float xe = x[e];
    Xh[row * DD + e] = (_Float16)xe;
    #pragma unroll
    for (int j = 0; j < RR; ++j) dj[j] += Wl[e * RR + j] * xe;
  }
  float acc = 0.f;
  #pragma unroll
  for (int j = 0; j < RR; ++j) acc += dj[j] * dj[j];

  red[tid] = acc;
  __syncthreads();
  for (int s = 128; s > 0; s >>= 1) {
    if (tid < s) red[tid] += red[tid + s];
    __syncthreads();
  }
  if (tid == 0) atomicAdd(&trace[b], red[0] * 0.011048543456039805f);
}

// ---------------------------------------------------------------------------
// Pass C: flash attention with WMMA f16->f32, ping-pong LDS double buffering,
// async global->LDS fills for the row-major key copy.
// Block = 256 threads (8 wave32), 128 query rows per block; grid (16, 8).
// ---------------------------------------------------------------------------
__global__ __launch_bounds__(256) void k_attn(const _Float16* __restrict__ Xh,
                                              const _Float16* __restrict__ Sh,
                                              const float* __restrict__ trace,
                                              float* __restrict__ out) {
  __shared__ __align__(16) unsigned char smem[SMEM_BYTES];
  _Float16* Qh = (_Float16*)smem;                 // prologue alias: [128][QH_LD]
  _Float16* Pb = (_Float16*)(smem + P_OFF);       // [8 waves][16][P_LD]

  const int tid  = threadIdx.x;
  const int wave = tid >> 5;
  const int lane = tid & 31;
  const int lm   = lane & 15;   // N / row-within-fragment
  const int g    = lane >> 4;   // half-wave group

  const int b  = blockIdx.y;
  const int q0 = blockIdx.x * 128;

  const float scale = 0.08838834764831845f;              // beta / sqrt(d)
  const float cdiag = (trace[b] * (1.0f / TT)) * scale;  // (tr/T)/sqrt(d)

  const _Float16* Xb = Xh + (size_t)b * TT * DD;

  // cooperative-load ownership for key tiles
  const int cp  = tid >> 3;        // key pair 0..31
  const int ccs = (tid & 7) * 16;  // dim segment

  // ---- Prologue: Q tile = Xq @ S (S symmetric: column j == row j) ----
  {
    const int qrow = q0 + wave * 16 + lm;
    const _Float16* arow = Xb + (size_t)qrow * DD;
    #pragma unroll
    for (int jt = 0; jt < 8; ++jt) {
      v8f acc = zero8();
      #pragma unroll
      for (int kc = 0; kc < 4; ++kc) {
        v16h a  = ld16(arow + kc * 32 + g * 8, arow + kc * 32 + 16 + g * 8);
        const _Float16* brow = Sh + (jt * 16 + lm) * DD + kc * 32 + g * 8;
        v16h bf = ld16(brow, brow + 16);
        acc = wmma_f16(a, bf, acc);
      }
      // C layout: (M = v + 8*g, N = lm)
      #pragma unroll
      for (int v = 0; v < 8; ++v)
        Qh[(wave * 16 + v + 8 * g) * QH_LD + jt * 16 + lm] = (_Float16)acc[v];
    }
  }
  __syncthreads();
  v16h qfrag[4];  // resident Q A-fragments (K = 128 in 4 chunks of 32)
  {
    const _Float16* qrowp = Qh + (wave * 16 + lm) * QH_LD;
    #pragma unroll
    for (int kc = 0; kc < 4; ++kc)
      qfrag[kc] = ld16(qrowp + kc * 32 + g * 8, qrowp + kc * 32 + 16 + g * 8);
  }
  __syncthreads();  // Qh region now reusable as key buffer 0

  float mrow[8], lrow[8];
  v8f oacc[8];
  #pragma unroll
  for (int v = 0; v < 8; ++v) { mrow[v] = -3.0e38f; lrow[v] = 0.f; }
  #pragma unroll
  for (int dt = 0; dt < 8; ++dt) oacc[dt] = zero8();

  const int myrowbase = q0 + wave * 16 + 8 * g;  // query row of C element v

  // fill buffer 0 with key tile 0 (async row-major + reg-staged transpose)
  {
    v8h st[4];
    tile_async_rowmajor((_Float16*)smem, Xb, 0, cp, ccs);
    tile_load_regs(st, Xb, 0, cp, ccs);
    tile_store_tr((_Float16*)(smem + 17408), st, cp, ccs);
    async_drain();
  }
  __syncthreads();

  for (int kt = 0; kt < TT / 64; ++kt) {
    const int k0  = kt * 64;
    const int cur = kt & 1;
    _Float16* Xk  = (_Float16*)(smem + cur * BUF_STRIDE);
    _Float16* XkT = (_Float16*)(smem + cur * BUF_STRIDE + 17408);

    // kick off NEXT tile: async LDS fill + register staging for transpose
    v8h st[4];
    const bool have_next = (kt + 1 < TT / 64);
    if (have_next) {
      _Float16* XkN = (_Float16*)(smem + (cur ^ 1) * BUF_STRIDE);
      tile_async_rowmajor(XkN, Xb, k0 + 64, cp, ccs);
      tile_load_regs(st, Xb, k0 + 64, cp, ccs);
      if (kt + 2 < TT / 64)
        __builtin_prefetch(Xb + (size_t)(k0 + 128 + 2 * cp) * DD + ccs, 0, 1);
    }

    // GEMM1: scores(16x64) = Q(16x128) @ K^T ; B[k][n] = Xk[n][k]
    v8f sacc[4];
    #pragma unroll
    for (int nt = 0; nt < 4; ++nt) sacc[nt] = zero8();
    #pragma unroll
    for (int nt = 0; nt < 4; ++nt) {
      const _Float16* krow = Xk + (nt * 16 + lm) * XK_LD;
      #pragma unroll
      for (int kc = 0; kc < 4; ++kc) {
        v16h bf = ld16(krow + kc * 32 + g * 8, krow + kc * 32 + 16 + g * 8);
        sacc[nt] = wmma_f16(qfrag[kc], bf, sacc[nt]);
      }
    }
    // scale + trace-centered diagonal correction
    #pragma unroll
    for (int nt = 0; nt < 4; ++nt) {
      #pragma unroll
      for (int v = 0; v < 8; ++v) {
        float s = sacc[nt][v] * scale;
        if (myrowbase + v == k0 + nt * 16 + lm) s -= cdiag;
        sacc[nt][v] = s;
      }
    }
    // online softmax (row reductions across the 16 lanes holding a row)
    #pragma unroll
    for (int v = 0; v < 8; ++v) {
      float mx = fmaxf(fmaxf(sacc[0][v], sacc[1][v]),
                       fmaxf(sacc[2][v], sacc[3][v]));
      #pragma unroll
      for (int off = 1; off < 16; off <<= 1)
        mx = fmaxf(mx, __shfl_xor(mx, off, 32));
      const float mnew  = fmaxf(mrow[v], mx);
      const float alpha = __expf(mrow[v] - mnew);
      mrow[v] = mnew;
      lrow[v] *= alpha;
      #pragma unroll
      for (int dt = 0; dt < 8; ++dt) oacc[dt][v] *= alpha;
      float rsum = 0.f;
      #pragma unroll
      for (int nt = 0; nt < 4; ++nt) {
        const float p = __expf(sacc[nt][v] - mnew);
        rsum += p;
        Pb[(wave * 16 + v + 8 * g) * P_LD + nt * 16 + lm] = (_Float16)p;
      }
      #pragma unroll
      for (int off = 1; off < 16; off <<= 1)
        rsum += __shfl_xor(rsum, off, 32);
      lrow[v] += rsum;
    }
    asm volatile("s_wait_dscnt 0" ::: "memory");  // own-wave P store->load

    // GEMM2: O(16x128) += P(16x64) @ V(64x128) ; B[k][d] = XkT[d][k]
    #pragma unroll
    for (int kc2 = 0; kc2 < 2; ++kc2) {
      const _Float16* prow = Pb + (wave * 16 + lm) * P_LD;
      v16h pa = ld16(prow + kc2 * 32 + g * 8, prow + kc2 * 32 + 16 + g * 8);
      #pragma unroll
      for (int dt = 0; dt < 8; ++dt) {
        const _Float16* vrow = XkT + (dt * 16 + lm) * XKT_LD + kc2 * 32 + g * 8;
        v16h bf = ld16(vrow, vrow + 16);
        oacc[dt] = wmma_f16(pa, bf, oacc[dt]);
      }
    }

    // commit transposed copy, drain async fills, one barrier per iteration
    if (have_next) {
      _Float16* XkTN = (_Float16*)(smem + (cur ^ 1) * BUF_STRIDE + 17408);
      tile_store_tr(XkTN, st, cp, ccs);
    }
    async_drain();
    __syncthreads();
  }

  // epilogue: normalize and write fp32 output
  #pragma unroll
  for (int v = 0; v < 8; ++v) {
    const float inv = 1.0f / lrow[v];
    const int row = myrowbase + v;
    float* orow = out + ((size_t)b * TT + row) * DD;
    #pragma unroll
    for (int dt = 0; dt < 8; ++dt)
      orow[dt * 16 + lm] = oacc[dt][v] * inv;
  }
}

// ---------------------------------------------------------------------------
extern "C" void kernel_launch(void* const* d_in, const int* in_sizes, int n_in,
                              void* d_out, int out_size, void* d_ws, size_t ws_size,
                              hipStream_t stream) {
  (void)in_sizes; (void)n_in; (void)out_size; (void)ws_size;
  const float* X = (const float*)d_in[0];  // (8, 2048, 128) fp32
  const float* W = (const float*)d_in[1];  // (128, 64) fp32
  float* out = (float*)d_out;              // (8, 2048, 128) fp32

  char* ws = (char*)d_ws;
  _Float16* Sh    = (_Float16*)(ws);           // 32768 B : S in f16
  float*    trace = (float*)(ws + 32768);      // 32 B    : per-batch trace
  _Float16* Xh    = (_Float16*)(ws + 32800);   // 4 MB    : X in f16

  k_prep_s<<<1, 256, 0, stream>>>(W, Sh, trace);
  k_prep_x<<<64, 256, 0, stream>>>(X, W, Xh, trace);
  k_attn<<<dim3(16, 8), 256, 0, stream>>>(Xh, Sh, trace, out);
}
